// CSWT_Unet_65446711656652
// MI455X (gfx1250) — matrix operand order repaired and checked by reading, not compiled, not run on hardware
//
#include <hip/hip_runtime.h>
#include <hip/hip_bf16.h>

typedef __attribute__((ext_vector_type(16))) _Float16 v16h;
typedef __attribute__((ext_vector_type(8)))  _Float16 v8h;
typedef __attribute__((ext_vector_type(8)))  float    v8f;

#define LQ    4096          // tokens per image (64*64)
#define CCH   384           // channels
#define MTOT  32768         // B * L
#define QKVN  1152
#define MLPH  1536

// ---------------------------------------------------------------------------
// window position -> global token row (b*4096 + l)
// branch 0: H_sp=64, W_sp=4  (w = b*16 + wb);  branch 1: H_sp=4, W_sp=64 (w = b*16 + hb)
__device__ __forceinline__ int win_row(int br, int w, int n) {
  int b = w >> 4, wi = w & 15, l;
  if (br == 0) { int hs = n >> 2, ws = n & 3;  l = hs * 64 + wi * 4 + ws; }
  else         { int hs = n >> 6, ws = n & 63; l = (wi * 4 + hs) * 64 + ws; }
  return b * LQ + l;
}

__device__ __forceinline__ v16h cat16(v8h lo, v8h hi) {
  return __builtin_shufflevector(lo, hi, 0,1,2,3,4,5,6,7,8,9,10,11,12,13,14,15);
}

// async global->LDS copy of 16 bytes per lane (CDNA5, tracked by ASYNCcnt)
__device__ __forceinline__ void async_b128(unsigned lds_addr, unsigned goff,
                                           unsigned long long base) {
  asm volatile("global_load_async_to_lds_b128 %0, %1, %2"
               :: "v"(lds_addr), "v"(goff), "s"(base) : "memory");
}
__device__ __forceinline__ unsigned lds_lo32(const void* p) {
  return (unsigned)(unsigned long long)p;
}

// ---------------------------------------------------------------------------
// elementwise f32 -> f16 cast (weights)
__global__ void cast_f16_kernel(const float* __restrict__ in, _Float16* __restrict__ out, int n) {
  int i = blockIdx.x * 256 + threadIdx.x;
  if (i < n) out[i] = (_Float16)in[i];
}

// ---------------------------------------------------------------------------
// LayerNorm over C=384, one block per row, writes f16
__global__ __launch_bounds__(128) void ln_kernel(const float* __restrict__ X,
    const float* __restrict__ gw, const float* __restrict__ gb,
    _Float16* __restrict__ out)
{
  __shared__ float s1[128], s2[128];
  int row = blockIdx.x, tid = threadIdx.x;
  const float* x = X + (size_t)row * CCH;
  float a0 = x[tid], a1 = x[tid + 128], a2 = x[tid + 256];
  s1[tid] = a0 + a1 + a2;
  s2[tid] = a0 * a0 + a1 * a1 + a2 * a2;
  __syncthreads();
  for (int st = 64; st > 0; st >>= 1) {
    if (tid < st) { s1[tid] += s1[tid + st]; s2[tid] += s2[tid + st]; }
    __syncthreads();
  }
  float mu  = s1[0] * (1.0f / CCH);
  float var = s2[0] * (1.0f / CCH) - mu * mu;
  float inv = rsqrtf(var + 1e-5f);
  _Float16* o = out + (size_t)row * CCH;
  o[tid]       = (_Float16)((a0 - mu) * inv * gw[tid]       + gb[tid]);
  o[tid + 128] = (_Float16)((a1 - mu) * inv * gw[tid + 128] + gb[tid + 128]);
  o[tid + 256] = (_Float16)((a2 - mu) * inv * gw[tid + 256] + gb[tid + 256]);
}

// ---------------------------------------------------------------------------
// WMMA GEMM with double-buffered async-LDS staging.
// out[M,N] = A[M,K](f16) * W[N,K]^T(f16) + bias, optional GELU,
// optional per-column scale (cols < scaleCols), optional f32 residual add.
// block = 128 threads (4 waves); block tile = 128x64; wave tile = 32x64
// (2 A-fragments reused across 4 B-fragments -> 8 WMMAs per k-step).
__global__ __launch_bounds__(128) void gemm_wmma_kernel(
    const _Float16* __restrict__ A, const _Float16* __restrict__ W,
    const float* __restrict__ bias, const float* __restrict__ res,
    float* __restrict__ outF, _Float16* __restrict__ outH,
    int M, int N, int K, int act, int scaleCols, float scaleVal)
{
  __shared__ _Float16 At[2][128 * 40];  // [row][k] rows padded to 40 halves
  __shared__ _Float16 Wt[2][64 * 40];   // [col][k]
  int tid = threadIdx.x;
  int lane = tid & 31, wave = tid >> 5;
  int g = lane >> 4, c16 = lane & 15;
  int mBlk = blockIdx.x * 128, n0 = blockIdx.y * 64;

  // staging roles: thread t moves its own A row (4 x 16B chunks) + 2 W chunks
  unsigned long long Ab = (unsigned long long)A;
  unsigned long long Wb = (unsigned long long)W;
  unsigned baseA = (unsigned)((size_t)(mBlk + tid) * K * 2);
  int wr = tid >> 1, wq = (tid & 1) * 2;          // W row, 16B-chunk index (0/2)
  unsigned baseW0 = (unsigned)(((size_t)(n0 + wr) * K + (wq)     * 8) * 2);
  unsigned baseW1 = (unsigned)(((size_t)(n0 + wr) * K + (wq + 1) * 8) * 2);

  unsigned lA[2], lW0[2], lW1[2];
  #pragma unroll
  for (int b = 0; b < 2; ++b) {
    lA[b]  = lds_lo32(&At[b][tid * 40]);
    lW0[b] = lds_lo32(&Wt[b][wr * 40 + (wq)     * 8]);
    lW1[b] = lds_lo32(&Wt[b][wr * 40 + (wq + 1) * 8]);
  }

  v8f acc[2][4];
  #pragma unroll
  for (int mt = 0; mt < 2; ++mt)
    #pragma unroll
    for (int j = 0; j < 4; ++j) acc[mt][j] = v8f{};

  int nk = K >> 5;

  // prologue: stage 0
  {
    async_b128(lA[0] + 0,  baseA + 0,  Ab);
    async_b128(lA[0] + 16, baseA + 16, Ab);
    async_b128(lA[0] + 32, baseA + 32, Ab);
    async_b128(lA[0] + 48, baseA + 48, Ab);
    async_b128(lW0[0], baseW0, Wb);
    async_b128(lW1[0], baseW1, Wb);
  }

  for (int s = 0; s < nk; ++s) {
    int buf = s & 1;
    if (s + 1 < nk) {
      unsigned kb = (unsigned)((s + 1) * 64);   // 32 halves = 64 bytes
      int nb = buf ^ 1;
      async_b128(lA[nb] + 0,  baseA + kb + 0,  Ab);
      async_b128(lA[nb] + 16, baseA + kb + 16, Ab);
      async_b128(lA[nb] + 32, baseA + kb + 32, Ab);
      async_b128(lA[nb] + 48, baseA + kb + 48, Ab);
      async_b128(lW0[nb], baseW0 + kb, Wb);
      async_b128(lW1[nb], baseW1 + kb, Wb);
      asm volatile("s_wait_asynccnt 0x6" ::: "memory");  // stage s done, s+1 in flight
    } else {
      asm volatile("s_wait_asynccnt 0x0" ::: "memory");
    }
    __syncthreads();

    const _Float16* ap0 = &At[buf][(wave * 32 + c16) * 40 + g * 8];
    const _Float16* ap1 = &At[buf][(wave * 32 + 16 + c16) * 40 + g * 8];
    v16h a0 = cat16(*(const v8h*)ap0, *(const v8h*)(ap0 + 16));
    v16h a1 = cat16(*(const v8h*)ap1, *(const v8h*)(ap1 + 16));
    #pragma unroll
    for (int j = 0; j < 4; ++j) {
      const _Float16* wp = &Wt[buf][(j * 16 + c16) * 40 + g * 8];
      v16h b = cat16(*(const v8h*)wp, *(const v8h*)(wp + 16));
      acc[0][j] = __builtin_amdgcn_wmma_f32_16x16x32_f16(
          false, a0, false, b, (short)0, acc[0][j], false, false);
      acc[1][j] = __builtin_amdgcn_wmma_f32_16x16x32_f16(
          false, a1, false, b, (short)0, acc[1][j], false, false);
    }
    __syncthreads();
  }

  #pragma unroll
  for (int mt = 0; mt < 2; ++mt) {
    int m0 = mBlk + wave * 32 + mt * 16;
    #pragma unroll
    for (int j = 0; j < 4; ++j) {
      #pragma unroll
      for (int r = 0; r < 8; ++r) {
        int grow = m0 + r + 8 * g;
        int col  = n0 + j * 16 + c16;
        float v = acc[mt][j][r] + bias[col];
        if (act == 1) v = 0.5f * v * (1.0f + erff(v * 0.70710678118654752f));
        if (col < scaleCols) v *= scaleVal;
        size_t oi = (size_t)grow * N + col;
        if (res)  v += res[oi];
        if (outF) outF[oi] = v;
        if (outH) outH[oi] = (_Float16)v;
      }
    }
  }
}

// ---------------------------------------------------------------------------
// LePE depthwise 3x3 SAME conv per window image, reading v-section of qkv_f16.
// lepe layout: [br][w(128)][c(192)][n(256)] f32
__global__ __launch_bounds__(256) void lepe_conv_kernel(
    const _Float16* __restrict__ qkv,
    const float* __restrict__ w0, const float* __restrict__ b0,
    const float* __restrict__ w1, const float* __restrict__ b1,
    float* __restrict__ lepe)
{
  int idx = blockIdx.x * 256 + threadIdx.x;
  int n  = idx & 255;
  int cw_ = (idx >> 8);
  int c  = cw_ % 192;
  int wv = (cw_ / 192) & 127;
  int br = idx / (256 * 192 * 128);
  const float* cw = (br ? w1 : w0) + c * 9;
  float acc = (br ? b1 : b0)[c];
  int Hs = br ? 4 : 64, Ws = br ? 64 : 4;
  int hs = n / Ws, ws = n % Ws;
  int vcol = 768 + br * 192 + c;
  #pragma unroll
  for (int dy = -1; dy <= 1; ++dy)
    #pragma unroll
    for (int dx = -1; dx <= 1; ++dx) {
      int y = hs + dy, xx = ws + dx;
      if (y < 0 || y >= Hs || xx < 0 || xx >= Ws) continue;
      int gl = win_row(br, wv, y * Ws + xx);
      acc += cw[(dy + 1) * 3 + (dx + 1)] * (float)qkv[(size_t)gl * QKVN + vcol];
    }
  lepe[idx] = acc;
}

// ---------------------------------------------------------------------------
// Windowed attention: one block (2 waves) per (branch, window, head, 32-row slab).
// K staged to LDS via async copies; V transposed via register->ds scatter.
// S = (q*scale) K^T via 16 WMMAs/wave, softmax with half-wave shuffles,
// P*V via 16 WMMAs/wave, + lepe, scatter to attened f16 [32768,384].
__global__ __launch_bounds__(64) void attn_kernel(const _Float16* __restrict__ qkv,
    const float* __restrict__ lepe, _Float16* __restrict__ attOut)
{
  __shared__ _Float16 Ks[256 * 40];      // K row-major  [n][32], stride 40
  __shared__ _Float16 Vs[32 * 264];      // V transposed [hd][n], stride 264
  __shared__ _Float16 Pb[2 * 16 * 264];  // per-wave P   [16][256], stride 264

  int t = blockIdx.x;
  int rb = t & 7;  t >>= 3;
  int h  = t % 6;  t /= 6;
  int w  = t & 127;
  int br = t >> 7;

  int tid = threadIdx.x;
  int lane = tid & 31, wave = tid >> 5;
  int g = lane >> 4, c16 = lane & 15;
  int qc = br * 192 + h * 32;
  unsigned long long Qb = (unsigned long long)qkv;

  // stage K via async global->LDS (gathered rows, 4x16B chunks per row),
  // stage V transposed via register->LDS scatter: 4 window rows per thread
  #pragma unroll
  for (int i = 0; i < 4; ++i) {
    int n  = tid * 4 + i;
    int gl = win_row(br, w, n);
    unsigned kgo = (unsigned)(((size_t)gl * QKVN + 384 + qc) * 2);
    #pragma unroll
    for (int q = 0; q < 4; ++q)
      async_b128(lds_lo32(&Ks[n * 40 + q * 8]), kgo + q * 16, Qb);
    const _Float16* vp = qkv + (size_t)gl * QKVN + 768 + qc;
    v8h r0 = *(const v8h*)(vp),      r1 = *(const v8h*)(vp + 8);
    v8h r2 = *(const v8h*)(vp + 16), r3 = *(const v8h*)(vp + 24);
    #pragma unroll
    for (int c = 0; c < 8; ++c) {
      Vs[(c)      * 264 + n] = r0[c];
      Vs[(c + 8)  * 264 + n] = r1[c];
      Vs[(c + 16) * 264 + n] = r2[c];
      Vs[(c + 24) * 264 + n] = r3[c];
    }
  }
  asm volatile("s_wait_asynccnt 0x0" ::: "memory");
  __syncthreads();

  int m0 = rb * 32 + wave * 16;

  // Q fragment (q already scaled by hd^-0.5 in QKV epilogue)
  int nq0 = m0 + c16;
  const _Float16* qp = qkv + (size_t)win_row(br, w, nq0) * QKVN + qc + g * 8;
  v16h afr = cat16(*(const v8h*)qp, *(const v8h*)(qp + 16));

  // S = Q K^T  (16 column tiles, K=32 in one WMMA each)
  v8f st[16];
  #pragma unroll
  for (int tt = 0; tt < 16; ++tt) {
    const _Float16* kp = &Ks[(tt * 16 + c16) * 40 + g * 8];
    v16h bfr = cat16(*(const v8h*)kp, *(const v8h*)(kp + 16));
    v8f z = {};
    st[tt] = __builtin_amdgcn_wmma_f32_16x16x32_f16(
        false, afr, false, bfr, (short)0, z, false, false);
  }

  // softmax over each row (row index = r + 8*g, data across 16 lanes of half-wave)
  float rinv[8];
  #pragma unroll
  for (int r = 0; r < 8; ++r) {
    float m = st[0][r];
    #pragma unroll
    for (int tt = 1; tt < 16; ++tt) m = fmaxf(m, st[tt][r]);
    for (int off = 1; off < 16; off <<= 1) m = fmaxf(m, __shfl_xor(m, off, 32));
    float s = 0.0f;
    #pragma unroll
    for (int tt = 0; tt < 16; ++tt) { float e = __expf(st[tt][r] - m); st[tt][r] = e; s += e; }
    for (int off = 1; off < 16; off <<= 1) s += __shfl_xor(s, off, 32);
    rinv[r] = 1.0f / s;
  }

  // write P (f16) to LDS in A-matrix-friendly row-major layout
  _Float16* pw = &Pb[wave * 16 * 264];
  #pragma unroll
  for (int tt = 0; tt < 16; ++tt)
    #pragma unroll
    for (int r = 0; r < 8; ++r)
      pw[(r + 8 * g) * 264 + tt * 16 + c16] = (_Float16)(st[tt][r] * rinv[r]);
  __syncthreads();

  // O = P V   (N=32 -> two 16-wide tiles, K=256 in 8 steps)
  v8f o0 = {}, o1 = {};
  #pragma unroll
  for (int k0 = 0; k0 < 256; k0 += 32) {
    const _Float16* pp = pw + c16 * 264 + k0 + g * 8;
    v16h a2 = cat16(*(const v8h*)pp, *(const v8h*)(pp + 16));
    const _Float16* vp0 = &Vs[(c16)      * 264 + k0 + g * 8];
    const _Float16* vp1 = &Vs[(16 + c16) * 264 + k0 + g * 8];
    v16h b0 = cat16(*(const v8h*)vp0, *(const v8h*)(vp0 + 16));
    v16h b1 = cat16(*(const v8h*)vp1, *(const v8h*)(vp1 + 16));
    o0 = __builtin_amdgcn_wmma_f32_16x16x32_f16(false, a2, false, b0, (short)0, o0, false, false);
    o1 = __builtin_amdgcn_wmma_f32_16x16x32_f16(false, a2, false, b1, (short)0, o1, false, false);
  }

  // epilogue: + lepe, scatter back to token layout
  int lbase = (br * 128 + w) * 192 + h * 32;
  #pragma unroll
  for (int r = 0; r < 8; ++r) {
    int nq = m0 + r + 8 * g;
    int gl = win_row(br, w, nq);
    float v0 = o0[r] + lepe[(size_t)(lbase + c16)      * 256 + nq];
    float v1 = o1[r] + lepe[(size_t)(lbase + 16 + c16) * 256 + nq];
    _Float16* op = attOut + (size_t)gl * CCH + qc;
    op[c16]      = (_Float16)v0;
    op[16 + c16] = (_Float16)v1;
  }
}

// ---------------------------------------------------------------------------
extern "C" void kernel_launch(void* const* d_in, const int* in_sizes, int n_in,
                              void* d_out, int out_size, void* d_ws, size_t ws_size,
                              hipStream_t stream) {
  const float* x       = (const float*)d_in[0];
  const float* ln1_w   = (const float*)d_in[1];
  const float* ln1_b   = (const float*)d_in[2];
  const float* qkv_w   = (const float*)d_in[3];
  const float* qkv_b   = (const float*)d_in[4];
  const float* lepe_w0 = (const float*)d_in[5];
  const float* lepe_b0 = (const float*)d_in[6];
  const float* lepe_w1 = (const float*)d_in[7];
  const float* lepe_b1 = (const float*)d_in[8];
  const float* proj_w  = (const float*)d_in[9];
  const float* proj_b  = (const float*)d_in[10];
  const float* ln2_w   = (const float*)d_in[11];
  const float* ln2_b   = (const float*)d_in[12];
  const float* fc1_w   = (const float*)d_in[13];
  const float* fc1_b   = (const float*)d_in[14];
  const float* fc2_w   = (const float*)d_in[15];
  const float* fc2_b   = (const float*)d_in[16];
  float* out = (float*)d_out;

  char* ws = (char*)d_ws;
  size_t off = 0;
  auto alloc = [&](size_t bytes) { char* p = ws + off; off += (bytes + 255) & ~(size_t)255; return p; };

  _Float16* qkvW_h = (_Float16*)alloc((size_t)QKVN * CCH * 2);
  _Float16* projW_h = (_Float16*)alloc((size_t)CCH * CCH * 2);
  _Float16* fc1W_h  = (_Float16*)alloc((size_t)MLPH * CCH * 2);
  _Float16* fc2W_h  = (_Float16*)alloc((size_t)CCH * MLPH * 2);
  _Float16* xn_h    = (_Float16*)alloc((size_t)MTOT * CCH * 2);   // LN1 out; reused for LN2 out
  _Float16* qkv_h   = (_Float16*)alloc((size_t)MTOT * QKVN * 2);
  float*    lepe_f  = (float*)   alloc((size_t)2 * 128 * 192 * 256 * 4);
  _Float16* att_h   = (_Float16*)alloc((size_t)MTOT * CCH * 2);
  float*    y1_f    = (float*)   alloc((size_t)MTOT * CCH * 4);
  _Float16* h1_h    = (_Float16*)alloc((size_t)MTOT * MLPH * 2);
  (void)ws_size;

  // 1) weight casts
  cast_f16_kernel<<<(QKVN * CCH + 255) / 256, 256, 0, stream>>>(qkv_w, qkvW_h, QKVN * CCH);
  cast_f16_kernel<<<(CCH * CCH + 255) / 256, 256, 0, stream>>>(proj_w, projW_h, CCH * CCH);
  cast_f16_kernel<<<(MLPH * CCH + 255) / 256, 256, 0, stream>>>(fc1_w, fc1W_h, MLPH * CCH);
  cast_f16_kernel<<<(CCH * MLPH + 255) / 256, 256, 0, stream>>>(fc2_w, fc2W_h, CCH * MLPH);

  // 2) LN1
  ln_kernel<<<MTOT, 128, 0, stream>>>(x, ln1_w, ln1_b, xn_h);

  // 3) QKV GEMM (q columns pre-scaled by 1/sqrt(32))
  gemm_wmma_kernel<<<dim3(MTOT / 128, QKVN / 64), 128, 0, stream>>>(
      xn_h, qkvW_h, qkv_b, nullptr, nullptr, qkv_h,
      MTOT, QKVN, CCH, 0, 384, 0.17677669529663687f);

  // 4) LePE depthwise conv
  lepe_conv_kernel<<<(2 * 128 * 192 * 256) / 256, 256, 0, stream>>>(
      qkv_h, lepe_w0, lepe_b0, lepe_w1, lepe_b1, lepe_f);

  // 5) windowed attention (2 branches * 128 windows * 6 heads * 8 row slabs)
  attn_kernel<<<2 * 128 * 6 * 8, 64, 0, stream>>>(qkv_h, lepe_f, att_h);

  // 6) proj GEMM + residual(x) -> y1 (f32)
  gemm_wmma_kernel<<<dim3(MTOT / 128, CCH / 64), 128, 0, stream>>>(
      att_h, projW_h, proj_b, x, y1_f, nullptr,
      MTOT, CCH, CCH, 0, 0, 1.0f);

  // 7) LN2 (reuse xn_h)
  ln_kernel<<<MTOT, 128, 0, stream>>>(y1_f, ln2_w, ln2_b, xn_h);

  // 8) fc1 GEMM + exact GELU -> h1 (f16)
  gemm_wmma_kernel<<<dim3(MTOT / 128, MLPH / 64), 128, 0, stream>>>(
      xn_h, fc1W_h, fc1_b, nullptr, nullptr, h1_h,
      MTOT, MLPH, CCH, 1, 0, 1.0f);

  // 9) fc2 GEMM + residual(y1) -> d_out (f32)
  gemm_wmma_kernel<<<dim3(MTOT / 128, CCH / 64), 128, 0, stream>>>(
      h1_h, fc2W_h, fc2_b, y1_f, out, nullptr,
      MTOT, CCH, MLPH, 0, 0, 1.0f);
}
